// Model_71975061946916
// MI455X (gfx1250) — matrix operand order repaired and checked by reading, not compiled, not run on hardware
//
#include <hip/hip_runtime.h>
#include <hip/hip_bf16.h>
#include <stdint.h>

// Problem constants (match reference)
#define VOCAB 100000
#define EDIM  512
#define HDIM  512
#define BATCH 256
#define SEQ   200
#define KTILES (HDIM / 32)   // 16  K-tiles of 32 (bf16 wmma K)
#define NTILES (HDIM / 16)   // 32  N-tiles of 16
#define BTILES (BATCH / 16)  // 16  M-tiles of 16 (batch rows)
#define NT_HALF (NTILES / 2) // 16  N-tiles per half of Whh

typedef __attribute__((ext_vector_type(16))) __bf16 v16bf;
typedef __attribute__((ext_vector_type(8)))  float  v8f;
typedef __attribute__((ext_vector_type(4)))  unsigned int v4u;
typedef __attribute__((ext_vector_type(8)))  int v8i;
typedef __attribute__((ext_vector_type(4)))  int v4i;

// ---------------- workspace layout (bytes) ----------------
static constexpr size_t SZ_WP    = (size_t)EDIM * HDIM * 2;        // 512 KB bf16
static constexpr size_t OFF_WXHP = 0;
static constexpr size_t OFF_WHHP = OFF_WXHP + SZ_WP;
static constexpr size_t OFF_BIAS = OFF_WHHP + SZ_WP;               // f32 [H]
static constexpr size_t OFF_XEP  = (OFF_BIAS + HDIM * 4 + 255) & ~(size_t)255;
static constexpr size_t SZ_XEP   = (size_t)SEQ * BATCH * EDIM * 2; // 52.4 MB bf16 A-frags
static constexpr size_t OFF_ZP   = OFF_XEP + SZ_XEP;
static constexpr size_t SZ_ZP    = (size_t)SEQ * BATCH * HDIM * 4; // 104.9 MB f32 C-frag blobs
static constexpr size_t OFF_HSUM = OFF_ZP + SZ_ZP;                 // f32 [B*H] C-frag blobs

// float -> bf16 round-to-nearest-even
__device__ __forceinline__ unsigned short f2bf(float f) {
  union { float f; uint32_t u; } v; v.f = f;
  uint32_t u = v.u;
  return (unsigned short)((u + 0x7FFFu + ((u >> 16) & 1u)) >> 16);
}

// A-fragment (16x32 bf16) element offset for value (row m, K k) inside one
// [KTILES][32 lanes][16 elems] buffer. Layout per ISA 7.12.2:
// lanes 0-15: K 0-7 then 16-23; lanes 16-31: K 8-15 then 24-31.
__device__ __forceinline__ int a_off(int m, int k) {
  int kt   = k >> 5;
  int kpos = k & 31;
  int hh   = (kpos >> 3) & 1;                     // lane half
  int j    = (kpos & 7) + (((kpos >> 4) & 1) << 3);
  int lane = m + (hh << 4);
  return ((kt << 5) + lane) * 16 + j;
}

// -------- K0: pack Wxh/Whh into bf16 B-fragment blobs, fold biases --------
// B-frag (32x16 bf16): lane = (n&15) + 16*((k>>3)&1), elem j = (k&7) + 8*((k>>4)&1),
// blob order [nt][kt][lane][16].
__global__ void k_pack_weights(const float* __restrict__ Wxh,
                               const float* __restrict__ Whh,
                               const float* __restrict__ bxh,
                               const float* __restrict__ bhh,
                               unsigned short* __restrict__ WxhP,
                               unsigned short* __restrict__ WhhP,
                               float* __restrict__ biasC) {
  int idx = blockIdx.x * blockDim.x + threadIdx.x;
  const int NW = EDIM * HDIM;
  if (idx < 2 * NW) {
    const float* src = (idx < NW) ? Wxh : Whh;
    unsigned short* dst = (idx < NW) ? WxhP : WhhP;
    int e = (idx < NW) ? idx : (idx - NW);
    int k = e >> 9, n = e & (HDIM - 1);
    int nt = n >> 4, kt = k >> 5;
    int lane = (n & 15) + (((k >> 3) & 1) << 4);
    int j = (k & 7) + (((k >> 4) & 1) << 3);
    dst[(((size_t)nt * KTILES + kt) * 32 + lane) * 16 + j] = f2bf(src[e]);
  } else if (idx < 2 * NW + HDIM) {
    int h = idx - 2 * NW;
    biasC[h] = bxh[h] + bhh[h];
  }
}

// -------- K1: embedding gather -> bf16 A-fragment blobs --------
// XeP blob order [s][bt][kt][lane][16]; one thread = one lane of one blob.
__global__ void k_gather_pack(const int* __restrict__ x,
                              const float* __restrict__ emb,
                              unsigned short* __restrict__ XeP) {
  int idx = blockIdx.x * blockDim.x + threadIdx.x;
  if (idx >= SEQ * BTILES * KTILES * 32) return;
  int lane = idx & 31;
  int kt   = (idx >> 5) & (KTILES - 1);
  int bt   = (idx >> 9) & (BTILES - 1);
  int s    = idx >> 13;
  int m = lane & 15, half = lane >> 4;
  int b = bt * 16 + m;
  int tok = x[b * SEQ + s];
  const float* src = emb + (size_t)tok * EDIM + kt * 32;
  unsigned short* dst =
      XeP + ((((size_t)s * BTILES + bt) * KTILES + kt) * 32 + lane) * 16;
#pragma unroll
  for (int j = 0; j < 8; ++j) dst[j]     = f2bf(src[half * 8 + j]);
#pragma unroll
  for (int j = 0; j < 8; ++j) dst[8 + j] = f2bf(src[16 + half * 8 + j]);
}

// -------- K2: Z = Xe @ Wxh + (bxh+bhh), stored as f32 C-frag blobs --------
// One wave = one 16x16 output tile; 8 waves / 256-thread block.
__global__ void k_gemm_xwxh(const unsigned short* __restrict__ XeP,
                            const unsigned short* __restrict__ WxhP,
                            const float* __restrict__ biasC,
                            float* __restrict__ ZP) {
  int wave = blockIdx.x * (blockDim.x >> 5) + (threadIdx.x >> 5);
  int lane = threadIdx.x & 31;
  int nt = wave & (NTILES - 1);
  int bt = (wave >> 5) & (BTILES - 1);
  int s  = wave >> 9;
  if (s >= SEQ) return;  // wave-uniform guard (grid is exact)
  const v16bf* A  = (const v16bf*)(XeP + (((size_t)s * BTILES + bt) * KTILES) * 32 * 16);
  const v16bf* Bm = (const v16bf*)(WxhP + ((size_t)nt * KTILES) * 32 * 16);
  v8f acc = {};
#pragma unroll
  for (int kt = 0; kt < KTILES; ++kt) {
    v16bf a = A[kt * 32 + lane];
    v16bf b = Bm[kt * 32 + lane];
    acc = __builtin_amdgcn_wmma_f32_16x16x32_bf16(false, a, false, b,
                                                  (short)0, acc, false, false);
  }
  float bv = biasC[nt * 16 + (lane & 15)];
#pragma unroll
  for (int i = 0; i < 8; ++i) acc[i] += bv;
  *(v8f*)(ZP + ((((size_t)s * BTILES + bt) * NTILES + nt) * 32 + lane) * 8) = acc;
}

// -------- K3: recurrence h' = relu(Z_t + h @ Whh), per 16-row batch tile ----
// 16 waves/WG. Whh split: N-tiles 0..15 register-resident B-frags (one per
// wave, 128 VGPRs), N-tiles 16..31 resident in LDS (256 KB, loaded once via
// the Tensor Data Mover). h tile double-buffered as bf16 A-frags in LDS
// (2 x 16 KB); one __syncthreads per step. LDS total = 288 KB <= 320 KB/WGP.
__global__ void __launch_bounds__(512, 1)
k_recurrence(const unsigned short* __restrict__ WhhP,
             const float* __restrict__ ZP,
             float* __restrict__ HsumP) {
  // WhhL declared first => assumed LDS offset 0 (used in TDM descriptor).
  __shared__ unsigned short WhhL[NT_HALF * KTILES * 32 * 16];  // 256 KB
  __shared__ unsigned short hA[2][KTILES * 32 * 16];           // 32 KB

  int lane = threadIdx.x & 31;
  int w    = threadIdx.x >> 5;   // 0..15
  int bt   = blockIdx.x;         // 0..15
  int nt0  = w;                  // register half (N-tiles 0..15)
  int nt1  = NT_HALF + w;        // LDS half      (N-tiles 16..31)

  // ---- TDM: DMA Whh upper half (256 KB, contiguous blob region) into LDS.
  // 2D descriptor: 4 lines x 32768 bf16 elems (64 KB), stride 32768 elems.
  if (threadIdx.x == 0) {
    uint64_t ga = (uint64_t)(uintptr_t)(WhhP + (size_t)NT_HALF * KTILES * 32 * 16);
    v4u g0;
    g0[0] = 1u;                                   // count=1, user mode
    g0[1] = 0u;                                   // lds_addr = offset of WhhL
    g0[2] = (unsigned int)ga;                     // global_addr[31:0]
    g0[3] = (unsigned int)((ga >> 32) & 0x1FFFFFFu) | (2u << 30);  // addr[56:32] | type=2
    v8i g1;
    g1[0] = (int)(1u << 16);                      // workgroup_mask=0, data_size=1 (2B)
    g1[1] = (int)0x80000000u;                     // tensor_dim0[15:0]=32768 -> bits 31:16
    g1[2] = (int)(4u << 16);                      // tensor_dim0[31:16]=0 | tensor_dim1[15:0]=4
    g1[3] = (int)0x80000000u;                     // tensor_dim1[31:16]=0 | tile_dim0=32768
    g1[4] = 4;                                    // tile_dim1=4 | tile_dim2=0
    g1[5] = 32768;                                // tensor_dim0_stride[31:0]
    g1[6] = 0;                                    // stride[47:32] | dim1_stride[15:0]
    g1[7] = 0;
    v4i g2 = {0, 0, 0, 0};
    v4i g3 = {0, 0, 0, 0};
    v8i g4 = {0, 0, 0, 0, 0, 0, 0, 0};            // unused extended group
    __builtin_amdgcn_tensor_load_to_lds(g0, g1, g2, g3, g4, 0);
    __builtin_amdgcn_s_wait_tensorcnt(0);
  }

  // ---- register-resident B fragments for N-tiles 0..15 (128 VGPRs/wave)
  v16bf B0[KTILES];
#pragma unroll
  for (int kt = 0; kt < KTILES; ++kt) {
    B0[kt] = *(const v16bf*)(WhhP + (((size_t)nt0 * KTILES + kt) * 32 + lane) * 16);
  }
  for (int i = threadIdx.x; i < KTILES * 32 * 16; i += 512) hA[0][i] = 0;

  __syncthreads();  // TDM data + hA[0] zeros visible to all waves

  v8f hs0 = {}, hs1 = {};
  for (int s = 0; s < SEQ; ++s) {
    int cur = s & 1, nxt = cur ^ 1;
    const float* zb = ZP + ((size_t)s * BTILES + bt) * NTILES * 32 * 8;
    v8f acc0 = *(const v8f*)(zb + ((size_t)nt0 * 32 + lane) * 8);
    v8f acc1 = *(const v8f*)(zb + ((size_t)nt1 * 32 + lane) * 8);
    if (s + 1 < SEQ)  // pull next step's Z blob toward the WGP
      __builtin_prefetch(
          (const void*)(ZP + ((size_t)(s + 1) * BTILES + bt) * NTILES * 32 * 8 +
                        ((size_t)nt0 * 32 + lane) * 8), 0, 3);
    if (s) __syncthreads();  // hA[cur] fully written by previous step
#pragma unroll
    for (int kt = 0; kt < KTILES; ++kt) {
      v16bf a  = *(const v16bf*)&hA[cur][(kt * 32 + lane) * 16];
      v16bf b1 = *(const v16bf*)&WhhL[(((size_t)w * KTILES + kt) * 32 + lane) * 16];
      acc0 = __builtin_amdgcn_wmma_f32_16x16x32_bf16(false, a, false, B0[kt],
                                                     (short)0, acc0, false, false);
      acc1 = __builtin_amdgcn_wmma_f32_16x16x32_bf16(false, a, false, b1,
                                                     (short)0, acc1, false, false);
    }
#pragma unroll
    for (int i = 0; i < 8; ++i) {
      acc0[i] = fmaxf(acc0[i], 0.0f);
      acc1[i] = fmaxf(acc1[i], 0.0f);
      hs0[i] += acc0[i];
      hs1[i] += acc1[i];
    }
    // scatter h' (C layout) into next A-frag buffer (cross-lane via LDS)
    int n0 = nt0 * 16 + (lane & 15);
    int n1 = nt1 * 16 + (lane & 15);
#pragma unroll
    for (int i = 0; i < 8; ++i) {
      int row = i + ((lane >> 4) << 3);
      hA[nxt][a_off(row, n0)] = f2bf(acc0[i]);
      hA[nxt][a_off(row, n1)] = f2bf(acc1[i]);
    }
  }
  *(v8f*)(HsumP + (((size_t)bt * NTILES + nt0) * 32 + lane) * 8) = hs0;
  *(v8f*)(HsumP + (((size_t)bt * NTILES + nt1) * 32 + lane) * 8) = hs1;
}

// -------- K4: out = (Hsum/S) @ Wfc + bfc  (tiny) --------
__global__ void k_final(const float* __restrict__ HsumP,
                        const float* __restrict__ Wfc,
                        const float* __restrict__ bfc,
                        float* __restrict__ out) {
  int idx = blockIdx.x * blockDim.x + threadIdx.x;
  if (idx >= BATCH * 4) return;
  int b = idx >> 2, o = idx & 3;
  int bt = b >> 4, mm = b & 15;
  int i = mm & 7, hl = mm >> 3;
  float sum = 0.0f;
  for (int h = 0; h < HDIM; ++h) {
    int nt = h >> 4, n = h & 15;
    int lanep = n + (hl << 4);
    sum += HsumP[(((size_t)bt * NTILES + nt) * 32 + lanep) * 8 + i] * Wfc[h * 4 + o];
  }
  out[idx] = sum * (1.0f / SEQ) + bfc[o];
}

extern "C" void kernel_launch(void* const* d_in, const int* in_sizes, int n_in,
                              void* d_out, int out_size, void* d_ws, size_t ws_size,
                              hipStream_t stream) {
  const int*   x   = (const int*)d_in[0];
  const float* emb = (const float*)d_in[1];
  const float* Wxh = (const float*)d_in[2];
  const float* bxh = (const float*)d_in[3];
  const float* Whh = (const float*)d_in[4];
  const float* bhh = (const float*)d_in[5];
  const float* Wfc = (const float*)d_in[6];
  const float* bfc = (const float*)d_in[7];
  char* ws = (char*)d_ws;
  unsigned short* WxhP  = (unsigned short*)(ws + OFF_WXHP);
  unsigned short* WhhP  = (unsigned short*)(ws + OFF_WHHP);
  float*          biasC = (float*)(ws + OFF_BIAS);
  unsigned short* XeP   = (unsigned short*)(ws + OFF_XEP);
  float*          ZP    = (float*)(ws + OFF_ZP);
  float*          HsumP = (float*)(ws + OFF_HSUM);

  k_pack_weights<<<(2 * EDIM * HDIM + HDIM + 255) / 256, 256, 0, stream>>>(
      Wxh, Whh, bxh, bhh, WxhP, WhhP, biasC);
  k_gather_pack<<<(SEQ * BTILES * KTILES * 32) / 256, 256, 0, stream>>>(x, emb, XeP);
  k_gemm_xwxh<<<(SEQ * BTILES * NTILES) / 8, 256, 0, stream>>>(XeP, WxhP, biasC, ZP);
  k_recurrence<<<BTILES, 512, 0, stream>>>(WhhP, ZP, HsumP);
  k_final<<<(BATCH * 4 + 255) / 256, 256, 0, stream>>>(HsumP, Wfc, bfc, (float*)d_out);
}
